// IntraSelfAttention_67516885893461
// MI455X (gfx1250) — matrix operand order repaired and checked by reading, not compiled, not run on hardware
//
#include <hip/hip_runtime.h>
#include <hip/hip_bf16.h>

// ---- CDNA5 (gfx1250) flash-attention-style kernel, wave32 + WMMA f16 ----
typedef _Float16 v16h __attribute__((ext_vector_type(16)));
typedef _Float16 v8h  __attribute__((ext_vector_type(8)));
typedef _Float16 v4h  __attribute__((ext_vector_type(4)));
typedef float    v8f  __attribute__((ext_vector_type(8)));
typedef float    v4f  __attribute__((ext_vector_type(4)));

#define BB 8
#define SS 1024
#define DD 768
#define EPS 1e-7f

static __device__ __forceinline__ v8f wmma_f16(v16h a, v16h b, v8f c) {
    return __builtin_amdgcn_wmma_f32_16x16x32_f16(false, a, false, b, (short)0, c,
                                                  false, false);
}
static __device__ __forceinline__ v16h cat8(v8h lo, v8h hi) {
    return __builtin_shufflevector(lo, hi, 0,1,2,3,4,5,6,7,8,9,10,11,12,13,14,15);
}

// ---------------- prepass: f32 -> f16, row-major copy ----------------
__global__ __launch_bounds__(256) void cvt16_kernel(const float* __restrict__ A,
                                                    _Float16* __restrict__ A16) {
    const size_t i = (size_t)blockIdx.x * 256 + threadIdx.x;   // float4 index
    v4f x = ((const v4f*)A)[i];
    v4h h;
    h[0] = (_Float16)x[0]; h[1] = (_Float16)x[1];
    h[2] = (_Float16)x[2]; h[3] = (_Float16)x[3];
    ((v4h*)A16)[i] = h;
}

// ------------- prepass: f32 -> f16 transposed copy A16T[b][d][s] -------------
// Strided reads hit L2 (A streamed through L2 by cvt16); writes are coalesced.
__global__ __launch_bounds__(256) void cvtT16_kernel(const float* __restrict__ A,
                                                     _Float16* __restrict__ A16T) {
    const size_t i  = (size_t)blockIdx.x * 256 + threadIdx.x;  // half4 index of A16T
    const int s4 = (int)(i & 255);           // 256 half4 per (b,d) row
    const int d  = (int)((i >> 8) % DD);
    const int b  = (int)(i / (DD * 256));
    const int s  = s4 * 4;
    const float* src = A + ((size_t)(b * SS + s)) * DD + d;
    v4h h;
    h[0] = (_Float16)src[0 * DD];
    h[1] = (_Float16)src[1 * DD];
    h[2] = (_Float16)src[2 * DD];
    h[3] = (_Float16)src[3 * DD];
    ((v4h*)A16T)[i] = h;
}

// ---------------- main attention kernel: 8 waves / 256 threads ----------------
// Block handles 16 query rows of one batch. Per iteration: 128 keys.
//   stage 1: wave w computes full 16x16 score tile for key-subtile w (24 wmma),
//            applies exp * colmask, parks e (f16) in LDS (C-layout -> A-layout).
//   stage 2: every wave does e @ V over its private 96-wide D-slice (24 wmma),
//            wave 0 also folds e @ ones into the row denominators (4 wmma).
__global__ __launch_bounds__(256) void
IntraSelfAttention_67516885893461_kernel(const _Float16* __restrict__ A16,
                                         const _Float16* __restrict__ A16T,
                                         const float* __restrict__ A,
                                         const int* __restrict__ mask,
                                         float* __restrict__ out) {
    __shared__ __align__(16) _Float16 ldsQ[16][DD];      // query tile f16 (24 KB)
    __shared__ __align__(16) _Float16 ldsE[8][16][16];   // e tiles, one per wave (4 KB)
    __shared__ __align__(16) float    ldsMrow[16];
    __shared__ __align__(16) float    ldsDen[16];

    const int tid  = threadIdx.x;
    const int wave = tid >> 5;
    const int lane = tid & 31;
    const int g    = lane >> 4;
    const int mln  = lane & 15;

    const int b  = blockIdx.x >> 6;
    const int m0 = (blockIdx.x & 63) << 4;

    // stage Q tile into LDS (f32 -> f16 once)
    #pragma unroll
    for (int i = 0; i < 12; ++i) {
        const int f  = tid + 256 * i;       // float4 index within 16x768
        const int m  = f / 192;
        const int c4 = f - m * 192;
        v4f x = *((const v4f*)(A + ((size_t)(b * SS + m0 + m)) * DD) + c4);
        v4h h;
        h[0] = (_Float16)x[0]; h[1] = (_Float16)x[1];
        h[2] = (_Float16)x[2]; h[3] = (_Float16)x[3];
        *(v4h*)&ldsQ[m][c4 * 4] = h;
    }
    if (tid < 16) ldsMrow[tid] = (float)mask[b * SS + m0 + tid];
    __syncthreads();

    v16h onesb;
    #pragma unroll
    for (int h = 0; h < 16; ++h) onesb[h] = (_Float16)1.0f;

    v8f acc[6] = {};   // this wave's 96-wide output slice (6 N-tiles)
    v8f dc = {};       // wave 0 only: e @ ones -> row denominators

    for (int it = 0; it < SS / 128; ++it) {
        const int t0   = it * 128;
        const int tkey = t0 + wave * 16;

        // ---- stage 1: full 16x16 score tile, reduction over D = 24 chunks ----
        const _Float16* krow = A16 + ((size_t)(b * SS + tkey + mln)) * DD;
        if (it + 1 < SS / 128)
            __builtin_prefetch(krow + 128 * DD, 0, 0);   // next key tile -> global_prefetch
        v8f sc = {};
        #pragma unroll
        for (int c = 0; c < 24; ++c) {
            // A-frag (Q): halves 0..7 -> K=8g+0..7 ; 8..15 -> K=16+8g+0..7
            v8h qlo = *(const v8h*)&ldsQ[mln][c * 32 + 8 * g];
            v8h qhi = *(const v8h*)&ldsQ[mln][c * 32 + 16 + 8 * g];
            // B-frag (K^T): B[k=d][n=t] = A16[t][d], halves 0..7 -> d = 8g+0..7 ; 8..15 -> d = 16+8g+0..7
            v8h blo = *(const v8h*)(krow + c * 32 + 8 * g);
            v8h bhi = *(const v8h*)(krow + c * 32 + 16 + 8 * g);
            sc = wmma_f16(cat8(qlo, qhi), cat8(blo, bhi), sc);
        }
        // e = exp(s) * colmask ; C layout: vgpr r = (row r+8g, col mln)
        const float cm = (float)mask[b * SS + tkey + mln];
        #pragma unroll
        for (int r = 0; r < 8; ++r)
            ldsE[wave][r + 8 * g][mln] = (_Float16)(__expf(sc[r]) * cm);
        __syncthreads();

        // ---- stage 2: acc += e @ V over this wave's D-slice ----
        #pragma unroll
        for (int p = 0; p < 4; ++p) {   // 4 K=32 pairs of e-subtiles
            v8h elo = *(const v8h*)&ldsE[2 * p][mln][8 * g];
            v8h ehi = *(const v8h*)&ldsE[2 * p + 1][mln][8 * g];
            v16h ef = cat8(elo, ehi);
            if (wave == 0) dc = wmma_f16(ef, onesb, dc);   // row sums (denominator)
            #pragma unroll
            for (int j = 0; j < 6; ++j) {
                const int n = wave * 96 + j * 16 + mln;
                // B-frag (V): B[k=t][n=d] = A16T[d][t], halves 0..7 -> t = 8g+0..7 ; 8..15 -> t = 16+8g+0..7
                const _Float16* vp = A16T + ((size_t)(b * DD + n)) * SS + t0 + p * 32 + 8 * g;
                v8h blo = *(const v8h*)vp;
                v8h bhi = *(const v8h*)(vp + 16);
                acc[j] = wmma_f16(ef, cat8(blo, bhi), acc[j]);
            }
        }
        __syncthreads();   // protect ldsE before next iteration overwrites
    }

    // ---- epilogue: out = mrow * acc / (mrow * den + eps) ----
    if (wave == 0 && mln == 0) {   // lanes 0 (rows 0-7) and 16 (rows 8-15), col 0
        #pragma unroll
        for (int r = 0; r < 8; ++r) ldsDen[r + 8 * g] = dc[r];
    }
    __syncthreads();

    float scale[8];
    #pragma unroll
    for (int r = 0; r < 8; ++r) {
        const int m = r + 8 * g;
        const float mr = ldsMrow[m];
        scale[r] = mr / (mr * ldsDen[m] + EPS);
    }
    #pragma unroll
    for (int j = 0; j < 6; ++j) {
        const int n = wave * 96 + j * 16 + mln;
        #pragma unroll
        for (int r = 0; r < 8; ++r)
            out[((size_t)(b * SS + m0 + r + 8 * g)) * DD + n] = acc[j][r] * scale[r];
    }
}

// ---------------- fallback (round-1 kernel): no workspace needed ----------------
__global__ __launch_bounds__(128) void
IntraSelfAttention_fallback_kernel(const float* __restrict__ A,
                                   const int* __restrict__ mask,
                                   float* __restrict__ out) {
    __shared__ __align__(16) _Float16 ldsKV[32][DD];
    __shared__ __align__(16) _Float16 ldsKVT[DD][32];
    __shared__ __align__(16) float    ldsS[16][32];
    __shared__ __align__(16) float    ldsM[32];
    __shared__ __align__(16) float    ldsMrow[16];
    __shared__ __align__(16) float    ldsDen[16];

    const int tid  = threadIdx.x;
    const int wave = tid >> 5;
    const int lane = tid & 31;
    const int g    = lane >> 4;
    const int mln  = lane & 15;
    const int b  = blockIdx.x >> 6;
    const int m0 = (blockIdx.x & 63) << 4;

    if (tid < 16) ldsMrow[tid] = (float)mask[b * SS + m0 + tid];

    v16h qfrag[6];
    {
        const float* qrow = A + ((size_t)(b * SS + m0 + mln)) * DD;
        #pragma unroll
        for (int j = 0; j < 6; ++j) {
            const int d0 = wave * 192 + j * 32;
            v4f a0 = *(const v4f*)(qrow + d0 + 8 * g);
            v4f a1 = *(const v4f*)(qrow + d0 + 8 * g + 4);
            v4f a2 = *(const v4f*)(qrow + d0 + 16 + 8 * g);
            v4f a3 = *(const v4f*)(qrow + d0 + 16 + 8 * g + 4);
            v16h q;
            #pragma unroll
            for (int h = 0; h < 4; ++h) {
                q[h] = (_Float16)a0[h];     q[4 + h] = (_Float16)a1[h];
                q[8 + h] = (_Float16)a2[h]; q[12 + h] = (_Float16)a3[h];
            }
            qfrag[j] = q;
        }
    }

    v8f acc[12] = {};
    float denom = 0.0f;

    for (int t0 = 0; t0 < SS; t0 += 32) {
        __syncthreads();
        #pragma unroll 4
        for (int i = 0; i < 48; ++i) {
            const int f  = tid + 128 * i;
            const int t  = f / 192;
            const int c4 = f - t * 192;
            const int d  = c4 * 4;
            v4f x = *((const v4f*)(A + ((size_t)(b * SS + t0 + t)) * DD) + c4);
            _Float16 h0 = (_Float16)x[0], h1 = (_Float16)x[1];
            _Float16 h2 = (_Float16)x[2], h3 = (_Float16)x[3];
            ldsKV[t][d + 0] = h0; ldsKV[t][d + 1] = h1;
            ldsKV[t][d + 2] = h2; ldsKV[t][d + 3] = h3;
            ldsKVT[d + 0][t] = h0; ldsKVT[d + 1][t] = h1;
            ldsKVT[d + 2][t] = h2; ldsKVT[d + 3][t] = h3;
        }
        if (tid < 32) ldsM[tid] = (float)mask[b * SS + t0 + tid];
        {
            float* ps = &ldsS[0][0];
            #pragma unroll
            for (int k = 0; k < 4; ++k) ps[tid + 128 * k] = 0.0f;
        }
        __syncthreads();

        v8f c0 = {}, c1 = {};
        #pragma unroll
        for (int j = 0; j < 6; ++j) {
            const int dk = wave * 192 + j * 32 + 8 * g;
            v8h b0lo = *(const v8h*)&ldsKV[mln][dk];
            v8h b0hi = *(const v8h*)&ldsKV[mln][dk + 16];
            v8h b1lo = *(const v8h*)&ldsKV[16 + mln][dk];
            v8h b1hi = *(const v8h*)&ldsKV[16 + mln][dk + 16];
            c0 = wmma_f16(qfrag[j], cat8(b0lo, b0hi), c0);
            c1 = wmma_f16(qfrag[j], cat8(b1lo, b1hi), c1);
        }
        #pragma unroll
        for (int r = 0; r < 8; ++r) {
            atomicAdd(&ldsS[r + 8 * g][mln],      c0[r]);
            atomicAdd(&ldsS[r + 8 * g][16 + mln], c1[r]);
        }
        __syncthreads();

        v16h ea;
        {
            const int k0 = 8 * g, k1 = 16 + 8 * g;
            v4f s0a = *(const v4f*)&ldsS[mln][k0];
            v4f s0b = *(const v4f*)&ldsS[mln][k0 + 4];
            v4f s1a = *(const v4f*)&ldsS[mln][k1];
            v4f s1b = *(const v4f*)&ldsS[mln][k1 + 4];
            v4f m0a = *(const v4f*)&ldsM[k0];
            v4f m0b = *(const v4f*)&ldsM[k0 + 4];
            v4f m1a = *(const v4f*)&ldsM[k1];
            v4f m1b = *(const v4f*)&ldsM[k1 + 4];
            float part = 0.0f;
            #pragma unroll
            for (int h = 0; h < 4; ++h) {
                float e0 = __expf(s0a[h]) * m0a[h];
                float e1 = __expf(s0b[h]) * m0b[h];
                float e2 = __expf(s1a[h]) * m1a[h];
                float e3 = __expf(s1b[h]) * m1b[h];
                ea[h] = (_Float16)e0;      ea[4 + h] = (_Float16)e1;
                ea[8 + h] = (_Float16)e2;  ea[12 + h] = (_Float16)e3;
                part += e0 + e1 + e2 + e3;
            }
            part += __shfl_xor(part, 16);
            denom += part;
        }
        #pragma unroll
        for (int j = 0; j < 12; ++j) {
            const int n  = wave * 192 + j * 16 + mln;
            const int kb = 8 * g;
            v8h blo = *(const v8h*)&ldsKVT[n][kb];
            v8h bhi = *(const v8h*)&ldsKVT[n][kb + 16];
            acc[j] = wmma_f16(ea, cat8(blo, bhi), acc[j]);
        }
    }

    if (wave == 0 && lane < 16) ldsDen[lane] = denom;
    __syncthreads();

    float scale[8];
    #pragma unroll
    for (int r = 0; r < 8; ++r) {
        const int m = r + 8 * g;
        const float mr = ldsMrow[m];
        scale[r] = mr / (mr * ldsDen[m] + EPS);
    }
    #pragma unroll
    for (int j = 0; j < 12; ++j) {
        const int n = wave * 192 + j * 16 + mln;
        #pragma unroll
        for (int r = 0; r < 8; ++r) {
            const int m = r + 8 * g;
            out[((size_t)(b * SS + m0 + m)) * DD + n] = acc[j][r] * scale[r];
        }
    }
}

extern "C" void kernel_launch(void* const* d_in, const int* in_sizes, int n_in,
                              void* d_out, int out_size, void* d_ws, size_t ws_size,
                              hipStream_t stream) {
    const float* A    = (const float*)d_in[0];
    const int*   mask = (const int*)d_in[1];
    float*       out  = (float*)d_out;
    (void)in_sizes; (void)n_in; (void)out_size;

    const size_t nElem = (size_t)BB * SS * DD;            // 6,291,456
    const size_t need  = 2 * nElem * sizeof(_Float16);    // ~25.2 MB workspace

    if (ws_size >= need) {
        _Float16* A16  = (_Float16*)d_ws;
        _Float16* A16T = A16 + nElem;
        const int cvtBlocks = (int)(nElem / 4 / 256);     // 6144
        cvt16_kernel <<<cvtBlocks, 256, 0, stream>>>(A, A16);
        cvtT16_kernel<<<cvtBlocks, 256, 0, stream>>>(A, A16T);
        IntraSelfAttention_67516885893461_kernel<<<BB * (SS / 16), 256, 0, stream>>>(
            A16, A16T, A, mask, out);
    } else {
        IntraSelfAttention_fallback_kernel<<<BB * (SS / 16), 128, 0, stream>>>(A, mask, out);
    }
}
